// PCT_23536420782441
// MI455X (gfx1250) — matrix operand order repaired and checked by reading, not compiled, not run on hardware
//
#include <hip/hip_runtime.h>
#include <hip/hip_bf16.h>
#include <math.h>

// ---------------------------------------------------------------------------
// CDNA5 WMMA types / helpers (wave32, V_WMMA_F32_16X16X32_BF16)
// ---------------------------------------------------------------------------
typedef __attribute__((ext_vector_type(16))) __bf16       bf16x16;
typedef __attribute__((ext_vector_type(8)))  float        f32x8;
typedef __attribute__((ext_vector_type(8)))  unsigned int u32x8;

// A-matrix (16x32, 16-bit): lane holds row M=lane%16; element e -> K index:
//   K = e + (lane<16 ? 0 : 16)
__device__ __forceinline__ int a_kidx(int e, int lane) {
  return e + ((lane & 16) ? 16 : 0);
}
// B-matrix (32x16, 16-bit): lane holds col N=lane%16; element e -> K index:
//   K = e + (lane<16 ? 0 : 16)
__device__ __forceinline__ int b_kidx(int e, int lane) {
  return e + ((lane & 16) ? 16 : 0);
}
// C/D (16x16 f32): element r -> row M = r + (lane<16?0:8); col N = lane%16

__device__ __forceinline__ float half16_max(float v) {
  v = fmaxf(v, __shfl_xor(v, 1, 32));
  v = fmaxf(v, __shfl_xor(v, 2, 32));
  v = fmaxf(v, __shfl_xor(v, 4, 32));
  v = fmaxf(v, __shfl_xor(v, 8, 32));
  return v;
}
__device__ __forceinline__ float half16_sum(float v) {
  v += __shfl_xor(v, 1, 32);
  v += __shfl_xor(v, 2, 32);
  v += __shfl_xor(v, 4, 32);
  v += __shfl_xor(v, 8, 32);
  return v;
}
__device__ __forceinline__ float wave32_sum(float v) {
  v = half16_sum(v);
  v += __shfl_xor(v, 16, 32);
  return v;
}
__device__ __forceinline__ float wave32_max(float v) {
  v = half16_max(v);
  v = fmaxf(v, __shfl_xor(v, 16, 32));
  return v;
}

// 16x16 energy tile: e[n,m] = sum_d XK[d, nrow0+n] * XK[d, m0+m]  (K=D loop)
// Loads batched into float arrays first so the compiler can issue all
// global_load_b32 before a single s_wait_loadcnt.
__device__ __forceinline__ f32x8 energy_tile(const float* __restrict__ K,
                                             int nrow0, int m0, int D, int N, int lane) {
  const int col = lane & 15;
  f32x8 e = {};
  for (int k0 = 0; k0 < D; k0 += 32) {
    float fa[16], fb[16];
#pragma unroll
    for (int q = 0; q < 16; ++q) {
      fa[q] = K[(size_t)(k0 + a_kidx(q, lane)) * N + nrow0 + col];
      fb[q] = K[(size_t)(k0 + b_kidx(q, lane)) * N + m0 + col];
    }
    bf16x16 a, bb;
#pragma unroll
    for (int q = 0; q < 16; ++q) { a[q] = (__bf16)fa[q]; bb[q] = (__bf16)fb[q]; }
    e = __builtin_amdgcn_wmma_f32_16x16x32_bf16(false, a, false, bb, (short)0, e, false, false);
  }
  return e;
}

// ---------------------------------------------------------------------------
// Generic fused 1x1-conv (channel GEMM) + bias + BN + activation + residual.
// Y[b, yOff+m, n] = act( scale[m] * (sum_k W[m,k]*(X[b,xOff+k,n] - Xs[..]) +
//                         bias2[b,m]) + shift[m] ) + R[b, rOff+m, n]
// Grid: (N/64, ceil(Cout/16), B), block 128 (4 waves, one 16x16 tile each).
// GUARD=false fast path: no conditionals in the K-loop; W tile staged in LDS
// as bf16 (row stride Cin+2 to avoid bank conflicts), read as packed b32.
// GUARD=true: clamped addresses + value selects (v_cndmask), loads never
// predicated -> no EXEC-mask branching.
// ---------------------------------------------------------------------------
template <bool GUARD, bool SUB>
__global__ void conv_bn_act(const float* __restrict__ X, int xBS, int xOff,
                            const float* __restrict__ Xs, int sBS, int sOff,
                            const float* __restrict__ W, int ldW,
                            const float* __restrict__ bias2,
                            const float* __restrict__ scale,
                            const float* __restrict__ shift,
                            const float* __restrict__ R, int rBS, int rOff,
                            float* __restrict__ Y, int yBS, int yOff,
                            int Cin, int Cout, int N, int act) {
  extern __shared__ char smem_raw[];
  __bf16* sw = reinterpret_cast<__bf16*>(smem_raw);

  const int lane = threadIdx.x & 31;
  const int wid  = threadIdx.x >> 5;
  const int n0   = (blockIdx.x * 4 + wid) * 16;
  const int m0   = blockIdx.y * 16;
  const int b    = blockIdx.z;
  const int col  = lane & 15;
  const int hi   = (lane & 16) ? 1 : 0;
  const int hioff = hi ? 8 : 0;
  const int CinP = Cin + 2;

  const float* Xb = X + ((size_t)b * xBS + xOff) * N;
  const float* Sb = SUB ? (Xs + ((size_t)b * sBS + sOff) * N) : nullptr;
  const int mrow  = m0 + col;  // A row handled by this lane

  if constexpr (!GUARD) {
    // cooperative coalesced preload of W tile rows m0..m0+15 (all K) as bf16
#pragma unroll 1
    for (int row = 0; row < 16; ++row)
      for (int k = threadIdx.x; k < Cin; k += 128)
        sw[row * CinP + k] = (__bf16)W[(size_t)(m0 + row) * ldW + k];
    __syncthreads();
  }

  f32x8 acc = {};
  for (int k0 = 0; k0 < Cin; k0 += 32) {
    bf16x16 a, bb;
    if constexpr (!GUARD) {
      // A from LDS: elements (q, q+1) are K-consecutive -> one b32 read each
      u32x8 aw;
#pragma unroll
      for (int q = 0; q < 16; q += 2) {
        int k = k0 + (hi ? 16 : 0) + q;
        aw[q >> 1] = *reinterpret_cast<const unsigned int*>(&sw[col * CinP + k]);
      }
      a = __builtin_bit_cast(bf16x16, aw);
    } else {
      float fa[16];
      const int mc = mrow < Cout ? mrow : 0;
#pragma unroll
      for (int q = 0; q < 16; ++q) {
        int ka = k0 + a_kidx(q, lane);
        int kc = ka < Cin ? ka : 0;
        float v = W[(size_t)mc * ldW + kc];
        fa[q] = (ka < Cin && mrow < Cout) ? v : 0.f;
      }
#pragma unroll
      for (int q = 0; q < 16; ++q) a[q] = (__bf16)fa[q];
    }

    float fb[16];
#pragma unroll
    for (int q = 0; q < 16; ++q) {
      int kb = k0 + b_kidx(q, lane);
      int kc = GUARD ? (kb < Cin ? kb : 0) : kb;
      float v = Xb[(size_t)kc * N + n0 + col];
      if constexpr (SUB) v -= Sb[(size_t)kc * N + n0 + col];
      if constexpr (GUARD) v = (kb < Cin) ? v : 0.f;
      fb[q] = v;
    }
#pragma unroll
    for (int q = 0; q < 16; ++q) bb[q] = (__bf16)fb[q];

    if (k0 + 32 < Cin)  // gfx1250 global_prefetch_b8 for next K-slab of X
      __builtin_prefetch(&Xb[(size_t)(k0 + 32) * N + n0 + col], 0, 0);

    acc = __builtin_amdgcn_wmma_f32_16x16x32_bf16(false, a, false, bb, (short)0, acc, false, false);
  }

  float* Yb = Y + ((size_t)b * yBS + yOff) * N;
  const float* Rb = R ? (R + ((size_t)b * rBS + rOff) * N) : nullptr;
#pragma unroll
  for (int r = 0; r < 8; ++r) {
    int m = m0 + r + hi * 8;
    if (m >= Cout) continue;
    float t = acc[r];
    if (bias2) t += bias2[(size_t)b * Cout + m];
    float s  = scale ? scale[m] : 1.f;
    float sh = shift ? shift[m] : 0.f;
    t = t * s + sh;
    if (act == 1)      t = t > 0.f ? t : 0.f;
    else if (act == 2) t = t >= 0.f ? t : 0.2f * t;
    if (Rb) t += Rb[(size_t)m * N + n0 + col];
    Yb[(size_t)m * N + n0 + col] = t;
  }
}

// ---------------------------------------------------------------------------
// Attention pass 1: online-softmax row stats of energy = XK^T * XK.
// One wave per 16-row tile; loops all m column tiles. Grid (N/64, B), block 128.
// ---------------------------------------------------------------------------
__global__ void attn_rowstats(const float* __restrict__ XK,
                              float* __restrict__ rmax, float* __restrict__ rsum,
                              int D, int N) {
  const int lane = threadIdx.x & 31;
  const int wid  = threadIdx.x >> 5;
  const int n0   = (blockIdx.x * 4 + wid) * 16;
  const int b    = blockIdx.y;
  const int hi   = (lane & 16) ? 1 : 0;
  const float* K = XK + (size_t)b * D * N;

  float runM[8], runS[8];
#pragma unroll
  for (int r = 0; r < 8; ++r) { runM[r] = -3.0e38f; runS[r] = 0.f; }

  for (int m0 = 0; m0 < N; m0 += 16) {
    f32x8 e = energy_tile(K, n0, m0, D, N, lane);
#pragma unroll
    for (int r = 0; r < 8; ++r) {
      float tmax = half16_max(e[r]);
      float newM = fmaxf(runM[r], tmax);
      float ts   = half16_sum(__expf(e[r] - newM));
      runS[r] = runS[r] * __expf(runM[r] - newM) + ts;
      runM[r] = newM;
    }
  }
  if ((lane & 15) == 0) {
#pragma unroll
    for (int r = 0; r < 8; ++r) {
      int n = n0 + r + hi * 8;
      rmax[(size_t)b * N + n] = runM[r];
      rsum[(size_t)b * N + n] = runS[r];
    }
  }
}

// ---------------------------------------------------------------------------
// Attention pass 2: per (b, 16-col tile m0): recompute energy tiles, form
// p = exp(e - rmax)/rsum in registers, accumulate colsum and XR = (XV * p)
// across all n (K-loop). Wave holds all D=128 output rows (8 WMMA acc tiles).
// Grid (N/16, B), block 32.
// ---------------------------------------------------------------------------
__global__ void attn_apply(const float* __restrict__ XK, const float* __restrict__ XV,
                           const float* __restrict__ rmax, const float* __restrict__ rsum,
                           float* __restrict__ XR, int D, int N) {
  const int lane = threadIdx.x & 31;
  const int m0   = blockIdx.x * 16;
  const int b    = blockIdx.y;
  const int col  = lane & 15;
  const int hi   = (lane & 16) ? 1 : 0;
  const float* K = XK + (size_t)b * D * N;
  const float* V = XV + (size_t)b * D * N;

  f32x8 acc[8];
#pragma unroll
  for (int dt = 0; dt < 8; ++dt) acc[dt] = (f32x8){};
  float colAcc = 0.f;

  for (int n0 = 0; n0 < N; n0 += 32) {
    // two 16x16 energy/prob tiles covering K rows n0..n0+31
    f32x8 t0 = energy_tile(K, n0,      m0, D, N, lane);
    f32x8 t1 = energy_tile(K, n0 + 16, m0, D, N, lane);
#pragma unroll
    for (int r = 0; r < 8; ++r) {
      int na = n0 + r + hi * 8;
      int nb = n0 + 16 + r + hi * 8;
      float pa = __expf(t0[r] - rmax[(size_t)b * N + na]) / rsum[(size_t)b * N + na];
      float pb = __expf(t1[r] - rmax[(size_t)b * N + nb]) / rsum[(size_t)b * N + nb];
      t0[r] = pa; t1[r] = pb;
      colAcc += pa + pb;
    }
    // D-layout (two 16x16 tiles) -> B-layout (32x16) via cross-half shuffles
    float x0[8], x1[8];
#pragma unroll
    for (int r = 0; r < 8; ++r) {
      x0[r] = __shfl_xor(t0[r], 16, 32);
      x1[r] = __shfl_xor(t1[r], 16, 32);
    }
    bf16x16 pB;
#pragma unroll
    for (int q = 0; q < 8; ++q) {
      float lo = hi ? x1[q] : t0[q];   // K = q      (+16 on hi lanes)
      float hv = hi ? t1[q] : x0[q];   // K = q + 8  (+16 on hi lanes)
      pB[q]     = (__bf16)lo;
      pB[q + 8] = (__bf16)hv;
    }
    // XR[d, m] += XV[d, n-block] * p[n-block, m] for all 8 d-tiles
#pragma unroll
    for (int dt = 0; dt < 8; ++dt) {
      float fa[16];
#pragma unroll
      for (int q = 0; q < 16; ++q)
        fa[q] = V[(size_t)(dt * 16 + col) * N + n0 + a_kidx(q, lane)];
      bf16x16 a;
#pragma unroll
      for (int q = 0; q < 16; ++q) a[q] = (__bf16)fa[q];
      acc[dt] = __builtin_amdgcn_wmma_f32_16x16x32_bf16(false, a, false, pB, (short)0, acc[dt], false, false);
    }
  }

  float cs  = colAcc + __shfl_xor(colAcc, 16, 32);   // full column sum over all n
  float inv = 1.f / (1e-9f + cs);
  float* O  = XR + (size_t)b * D * N;
#pragma unroll
  for (int dt = 0; dt < 8; ++dt) {
#pragma unroll
    for (int r = 0; r < 8; ++r) {
      int d = dt * 16 + r + hi * 8;
      O[(size_t)d * N + m0 + col] = acc[dt][r] * inv;
    }
  }
}

// ---------------------------------------------------------------------------
// Global max/avg pool over N. One wave per (b,c). Grid = B*C/4 blocks of 128.
// ---------------------------------------------------------------------------
__global__ void pool_maxavg(const float* __restrict__ F, float* __restrict__ xmax,
                            float* __restrict__ xavg, int C, int N) {
  const int lane = threadIdx.x & 31;
  const int gw   = blockIdx.x * 4 + (threadIdx.x >> 5);
  const int b = gw / C, c = gw % C;
  const float* p = F + ((size_t)b * C + c) * N;
  float mx = -3.0e38f, sm = 0.f;
  for (int n = lane; n < N; n += 32) { float v = p[n]; mx = fmaxf(mx, v); sm += v; }
  mx = wave32_max(mx);
  sm = wave32_sum(sm);
  if (lane == 0) { xmax[(size_t)b * C + c] = mx; xavg[(size_t)b * C + c] = sm / (float)N; }
}

// P[b,o] = bs1[o] + sum_c ws1[o, F+c]*xmax[b,c] + ws1[o, 2F+c]*xavg[b,c]
__global__ void pool_proj(const float* __restrict__ ws1, const float* __restrict__ bs1,
                          const float* __restrict__ xmax, const float* __restrict__ xavg,
                          float* __restrict__ P, int F, int H) {
  const int lane = threadIdx.x & 31;
  const int gw   = blockIdx.x * 4 + (threadIdx.x >> 5);
  const int b = gw / H, o = gw % H;
  const float* Wm = ws1 + (size_t)o * 3 * F + F;
  const float* Wa = ws1 + (size_t)o * 3 * F + 2 * F;
  float s = 0.f;
  for (int c = lane; c < F; c += 32)
    s += Wm[c] * xmax[(size_t)b * F + c] + Wa[c] * xavg[(size_t)b * F + c];
  s = wave32_sum(s);
  if (lane == 0) P[(size_t)b * H + o] = s + bs1[o];
}

// Per-channel BN fold: scale = g/sqrt(v+eps); shift = b + scale*(bias - m).
// g==null -> scale=1, shift=bias (pure conv-bias layer).
__global__ void prep_ss(float* __restrict__ sc, float* __restrict__ sh,
                        const float* __restrict__ g, const float* __restrict__ be,
                        const float* __restrict__ mu, const float* __restrict__ va,
                        const float* __restrict__ bias, int C) {
  int i = blockIdx.x * blockDim.x + threadIdx.x;
  if (i >= C) return;
  float s, t;
  if (g) {
    s = g[i] * rsqrtf(va[i] + 1e-5f);
    t = be[i] + s * ((bias ? bias[i] : 0.f) - mu[i]);
  } else {
    s = 1.f;
    t = bias ? bias[i] : 0.f;
  }
  sc[i] = s; sh[i] = t;
}

// ---------------------------------------------------------------------------
// Host orchestration
// ---------------------------------------------------------------------------
static void launch_conv(hipStream_t s, const float* X, int xBS, int xOff,
                        const float* Xs, int sBS, int sOff,
                        const float* W, int ldW, const float* bias2,
                        const float* sc, const float* sh,
                        const float* R, int rBS, int rOff,
                        float* Y, int yBS, int yOff,
                        int B, int Cin, int Cout, int N, int act) {
  dim3 block(128);
  dim3 grid(N / 64, (Cout + 15) / 16, B);
  const bool guard = (Cin % 32 != 0) || (Cout % 16 != 0);
  const size_t smem = guard ? 0 : (size_t)16 * (Cin + 2) * sizeof(__bf16);
  if (guard) {
    if (Xs) conv_bn_act<true, true><<<grid, block, 0, s>>>(X, xBS, xOff, Xs, sBS, sOff, W, ldW,
                bias2, sc, sh, R, rBS, rOff, Y, yBS, yOff, Cin, Cout, N, act);
    else    conv_bn_act<true, false><<<grid, block, 0, s>>>(X, xBS, xOff, Xs, sBS, sOff, W, ldW,
                bias2, sc, sh, R, rBS, rOff, Y, yBS, yOff, Cin, Cout, N, act);
  } else {
    if (Xs) conv_bn_act<false, true><<<grid, block, smem, s>>>(X, xBS, xOff, Xs, sBS, sOff, W, ldW,
                bias2, sc, sh, R, rBS, rOff, Y, yBS, yOff, Cin, Cout, N, act);
    else    conv_bn_act<false, false><<<grid, block, smem, s>>>(X, xBS, xOff, Xs, sBS, sOff, W, ldW,
                bias2, sc, sh, R, rBS, rOff, Y, yBS, yOff, Cin, Cout, N, act);
  }
}

extern "C" void kernel_launch(void* const* d_in, const int* in_sizes, int n_in,
                              void* d_out, int out_size, void* d_ws, size_t ws_size,
                              hipStream_t stream) {
  const int B = 16, C0 = 3, N = 2048, D = 128, F = 1024, H1 = 512, OUT = 50;

  const float* x     = (const float*)d_in[0];
  const float* w1    = (const float*)d_in[1];
  const float* w2    = (const float*)d_in[2];
  const float* bn1g  = (const float*)d_in[3];  const float* bn1b = (const float*)d_in[4];
  const float* bn1m  = (const float*)d_in[5];  const float* bn1v = (const float*)d_in[6];
  const float* bn2g  = (const float*)d_in[7];  const float* bn2b = (const float*)d_in[8];
  const float* bn2m  = (const float*)d_in[9];  const float* bn2v = (const float*)d_in[10];
  const float* wqk   = (const float*)d_in[11];
  const float* wv    = (const float*)d_in[12]; const float* bv   = (const float*)d_in[13];
  const float* wt    = (const float*)d_in[14]; const float* bt   = (const float*)d_in[15];
  const float* sag   = (const float*)d_in[16]; const float* sab  = (const float*)d_in[17];
  const float* sam   = (const float*)d_in[18]; const float* sav  = (const float*)d_in[19];
  const float* wf    = (const float*)d_in[20];
  const float* bnfg  = (const float*)d_in[21]; const float* bnfb = (const float*)d_in[22];
  const float* bnfm  = (const float*)d_in[23]; const float* bnfv = (const float*)d_in[24];
  const float* ws1   = (const float*)d_in[25]; const float* bs1  = (const float*)d_in[26];
  const float* bs1g  = (const float*)d_in[27]; const float* bs1b = (const float*)d_in[28];
  const float* bs1m  = (const float*)d_in[29]; const float* bs1v = (const float*)d_in[30];
  const float* ws2   = (const float*)d_in[31]; const float* bs2  = (const float*)d_in[32];
  const float* bs2g  = (const float*)d_in[33]; const float* bs2b = (const float*)d_in[34];
  const float* bs2m  = (const float*)d_in[35]; const float* bs2v = (const float*)d_in[36];

  float* ws = (float*)d_ws;
  // workspace layout (floats)
  const size_t BUF  = (size_t)B * D * N;          // 4,194,304
  float* A_buf  = ws;                             // conv1 out; later reused as XR
  float* HP     = ws + BUF;                       // SA input (conv2 out)
  float* XK     = ws + 2 * BUF;
  float* XV     = ws + 3 * BUF;
  float* XR     = A_buf;                          // reuse: A dead after conv2
  float* CAT    = ws + 4 * BUF;                   // [B,512,N]; later reused as S1
  float* FUSED  = CAT + (size_t)B * 4 * D * N;    // [B,1024,N]
  float* SMALL  = FUSED + (size_t)B * F * N;
  float* RMAX   = SMALL;                 // B*N
  float* RSUM   = RMAX + (size_t)B * N;  // B*N
  float* XMAX   = RSUM + (size_t)B * N;  // B*F
  float* XAVG   = XMAX + (size_t)B * F;  // B*F
  float* PBIAS  = XAVG + (size_t)B * F;  // B*H1
  float* SS     = PBIAS + (size_t)B * H1;
  // SS sub-offsets
  float* bn1_sc = SS + 0;    float* bn1_sh = SS + 128;
  float* bn2_sc = SS + 256;  float* bn2_sh = SS + 384;
  float* bv_sc  = SS + 512;  // 4 * (sc 128 + sh 128)
  float* sa_sc  = SS + 1536; // 4 * (sc 128 + sh 128)
  float* bnf_sc = SS + 2560; float* bnf_sh = SS + 3584;
  float* s1_sc  = SS + 4608; float* s1_sh  = SS + 5120;
  float* s2_sc  = SS + 5632; float* s2_sh  = SS + 5696;
  float* S1     = CAT;                            // reuse CAT after wf conv

  // --- fold BN / bias params ---
  prep_ss<<<1, 256, 0, stream>>>(bn1_sc, bn1_sh, bn1g, bn1b, bn1m, bn1v, nullptr, D);
  prep_ss<<<1, 256, 0, stream>>>(bn2_sc, bn2_sh, bn2g, bn2b, bn2m, bn2v, nullptr, D);
  for (int i = 0; i < 4; ++i) {
    prep_ss<<<1, 256, 0, stream>>>(bv_sc + i * 256, bv_sc + i * 256 + 128,
                                   nullptr, nullptr, nullptr, nullptr, bv + i * D, D);
    prep_ss<<<1, 256, 0, stream>>>(sa_sc + i * 256, sa_sc + i * 256 + 128,
                                   sag + i * D, sab + i * D, sam + i * D, sav + i * D,
                                   bt + i * D, D);
  }
  prep_ss<<<4, 256, 0, stream>>>(bnf_sc, bnf_sh, bnfg, bnfb, bnfm, bnfv, nullptr, F);
  prep_ss<<<2, 256, 0, stream>>>(s1_sc, s1_sh, bs1g, bs1b, bs1m, bs1v, nullptr, H1);
  prep_ss<<<1, 256, 0, stream>>>(s2_sc, s2_sh, bs2g, bs2b, bs2m, bs2v, bs2, OUT);

  // --- embedding: h = relu(bn2(w2 @ relu(bn1(w1 @ x)))) ---
  launch_conv(stream, x, C0, 0, nullptr, 0, 0, w1, C0, nullptr, bn1_sc, bn1_sh,
              nullptr, 0, 0, A_buf, D, 0, B, C0, D, N, 1);
  launch_conv(stream, A_buf, D, 0, nullptr, 0, 0, w2, D, nullptr, bn2_sc, bn2_sh,
              nullptr, 0, 0, HP, D, 0, B, D, D, N, 1);

  // --- 4 offset self-attention layers; outputs concatenated into CAT ---
  for (int i = 0; i < 4; ++i) {
    const float* Hin = (i == 0) ? HP : CAT;
    int hBS  = (i == 0) ? D : 4 * D;
    int hOff = (i == 0) ? 0 : D * (i - 1);
    // xk = wqk @ h ; xv = wv @ h + bv
    launch_conv(stream, Hin, hBS, hOff, nullptr, 0, 0, wqk + (size_t)i * D * D, D,
                nullptr, nullptr, nullptr, nullptr, 0, 0, XK, D, 0, B, D, D, N, 0);
    launch_conv(stream, Hin, hBS, hOff, nullptr, 0, 0, wv + (size_t)i * D * D, D,
                nullptr, bv_sc + i * 256, bv_sc + i * 256 + 128,
                nullptr, 0, 0, XV, D, 0, B, D, D, N, 0);
    // flash-style two-pass attention (energy = xk^T xk, never materialized)
    attn_rowstats<<<dim3(N / 64, B), 128, 0, stream>>>(XK, RMAX, RSUM, D, N);
    attn_apply<<<dim3(N / 16, B), 32, 0, stream>>>(XK, XV, RMAX, RSUM, XR, D, N);
    // h = h + relu(bn(wt @ (h - x_r) + bt))  -> CAT slice i
    launch_conv(stream, Hin, hBS, hOff, XR, D, 0, wt + (size_t)i * D * D, D,
                nullptr, sa_sc + i * 256, sa_sc + i * 256 + 128,
                Hin, hBS, hOff, CAT, 4 * D, D * i, B, D, D, N, 1);
  }

  // --- fused = leakyrelu0.2(bn(wf @ cat)) ---
  launch_conv(stream, CAT, 4 * D, 0, nullptr, 0, 0, wf, 4 * D, nullptr, bnf_sc, bnf_sh,
              nullptr, 0, 0, FUSED, F, 0, B, 4 * D, F, N, 2);

  // --- global pools; fold broadcast halves of ws1 into per-(b,o) bias ---
  pool_maxavg<<<dim3((B * F) / 4), 128, 0, stream>>>(FUSED, XMAX, XAVG, F, N);
  pool_proj<<<dim3((B * H1) / 4), 128, 0, stream>>>(ws1, bs1, XMAX, XAVG, PBIAS, F, H1);

  // --- h = relu(bn(ws1[:, :F] @ fused + P)) ---   (S1 reuses CAT storage)
  launch_conv(stream, FUSED, F, 0, nullptr, 0, 0, ws1, 3 * F, PBIAS, s1_sc, s1_sh,
              nullptr, 0, 0, S1, H1, 0, B, F, H1, N, 1);

  // --- out = relu(bn(ws2 @ h + bs2)) ---
  launch_conv(stream, S1, H1, 0, nullptr, 0, 0, ws2, H1, nullptr, s2_sc, s2_sh,
              nullptr, 0, 0, (float*)d_out, OUT, 0, B, H1, OUT, N, 1);
}
